// AdjGkNO_32719060861288
// MI455X (gfx1250) — matrix operand order repaired and checked by reading, not compiled, not run on hardware
//
#include <hip/hip_runtime.h>
#include <hip/hip_bf16.h>
#include <math.h>

// Problem constants (match reference)
#define SDIM 65536
#define CDIM 64
#define MDIM 64
#define BDIM 8
#define FCDIM 128
#define NLAYERS 4
#define NCHUNK 32
#define CHUNK (SDIM / NCHUNK)   // 2048
#define STILES (SDIM / 64)      // 1024

typedef __attribute__((ext_vector_type(16))) __bf16 v16bf;
typedef __attribute__((ext_vector_type(8)))  float  v8f;
typedef __attribute__((ext_vector_type(4))) unsigned int tdm_v4u;
typedef __attribute__((ext_vector_type(8))) int          tdm_v8i;

union V16U { v16bf v; uint4 q[2]; };

__device__ __forceinline__ unsigned short f2bf(float f) {
  unsigned int u = __float_as_uint(f);
  u += 0x7FFFu + ((u >> 16) & 1u);          // round-to-nearest-even
  return (unsigned short)(u >> 16);
}
__device__ __forceinline__ float bf2f(unsigned short h) {
  return __uint_as_float(((unsigned int)h) << 16);
}
__device__ __forceinline__ float gelu_exact(float x) {
  return 0.5f * x * (1.0f + erff(x * 0.70710678118654752f));
}

// ---------------------------------------------------------------------------
// Tensor Data Mover: async 2D tile (rows x cols of bf16) global -> LDS.
// D# packing per CDNA5 ISA (2-group form, <=2D).
// pad_* : optional LDS row padding (pad_interval/pad_amount encodings per ISA).
__device__ __forceinline__ void tdm_load_2d(unsigned lds_off, const void* gptr,
                                            unsigned tile_d0, unsigned tile_d1,
                                            unsigned tensor_d0, unsigned tensor_d1,
                                            unsigned long long stride0_elems,
                                            unsigned pad_en, unsigned pad_interval,
                                            unsigned pad_amount) {
  unsigned long long ga = (unsigned long long)(uintptr_t)gptr;
  tdm_v4u g0;
  g0.x = 1u;                                                   // count=1, user D#
  g0.y = lds_off;                                              // lds_addr (bytes)
  g0.z = (unsigned)(ga & 0xFFFFFFFFu);                         // global_addr[31:0]
  g0.w = (unsigned)((ga >> 32) & 0x1FFFFFFu) | (2u << 30);     // addr[56:32] | type=2
  tdm_v8i g1;
  g1[0] = (int)((1u << 16) | (pad_en << 20) | (pad_interval << 22) | (pad_amount << 25));
  g1[1] = (int)((tensor_d0 & 0xFFFFu) << 16);                  // barrier_addr=0 | td0.lo
  g1[2] = (int)(((tensor_d0 >> 16) & 0xFFFFu) | ((tensor_d1 & 0xFFFFu) << 16));
  g1[3] = (int)(((tensor_d1 >> 16) & 0xFFFFu) | ((tile_d0 & 0xFFFFu) << 16));
  g1[4] = (int)(tile_d1 & 0xFFFFu);                            // tile_dim1 | tile_dim2=0
  g1[5] = (int)(stride0_elems & 0xFFFFFFFFull);                // tensor_dim0_stride lo
  g1[6] = (int)((stride0_elems >> 32) & 0xFFFFull);            // stride0 hi | stride1=0
  g1[7] = 0;
  asm volatile("tensor_load_to_lds %0, %1" :: "s"(g0), "s"(g1) : "memory");
}

// ---------------------------------------------------------------------------
// A operand, 16x32 bf16, from row-major [row][k] storage (ld in elements).
__device__ __forceinline__ v16bf frag_a(const unsigned short* base, int row0, int k0, int ld) {
  int lane = threadIdx.x & 31;
  int r  = row0 + (lane & 15);
  int kk = k0 + ((lane >> 4) << 3);
  V16U f;
  f.q[0] = *(const uint4*)(base + (size_t)r * ld + kk);
  f.q[1] = *(const uint4*)(base + (size_t)r * ld + kk + 16);
  return f.v;
}

// B operand, 32x16 bf16, from row-major Bᵀ storage [n][k] (ld in elements).
__device__ __forceinline__ v16bf frag_b(const unsigned short* baseT, int n0, int k0, int ld) {
  int lane = threadIdx.x & 31;
  int n  = n0 + (lane & 15);
  int kk = k0 + ((lane >> 4) << 3);
  V16U f;
  f.q[0] = *(const uint4*)(baseT + (size_t)n * ld + kk);
  f.q[1] = *(const uint4*)(baseT + (size_t)n * ld + kk + 16);
  return f.v;
}

__device__ __forceinline__ v8f wmma_bf16(v16bf a, v16bf b, v8f c) {
  return __builtin_amdgcn_wmma_f32_16x16x32_bf16(false, a, false, b, (short)0, c, false, false);
}

// C/D 16x16 f32 fragment -> row-major f32 [row][col]
__device__ __forceinline__ void store_c_f32(float* base, int row0, int col0, int ld, v8f acc) {
  int lane = threadIdx.x & 31;
  int n  = col0 + (lane & 15);
  int r0 = row0 + ((lane >> 4) << 3);
#pragma unroll
  for (int j = 0; j < 8; ++j) base[(size_t)(r0 + j) * ld + n] = acc[j];
}

// ---------------------------------------------------------------------------
// Prep: transpose+convert small weights. wT[l][o][c], fc1T[f][c]
__global__ void prep_weights_kernel(const float* w0, const float* w1, const float* w2,
                                    const float* w3, const float* fc1w,
                                    unsigned short* wT, unsigned short* fc1T) {
  const float* ww[4] = {w0, w1, w2, w3};
  int total = NLAYERS * CDIM * CDIM + FCDIM * CDIM;
  for (int idx = blockIdx.x * blockDim.x + threadIdx.x; idx < total; idx += gridDim.x * blockDim.x) {
    if (idx < NLAYERS * CDIM * CDIM) {
      int l = idx / (CDIM * CDIM);
      int r = idx - l * CDIM * CDIM;
      int o = r >> 6, c = r & 63;
      wT[l * CDIM * CDIM + o * CDIM + c] = f2bf(ww[l][c * CDIM + o]);   // w[c][o] -> wT[o][c]
    } else {
      int j = idx - NLAYERS * CDIM * CDIM;
      int f = j >> 6, c = j & 63;
      fc1T[f * CDIM + c] = f2bf(fc1w[c * FCDIM + f]);                   // fc1[c][f] -> fc1T[f][c]
    }
  }
}

// Prep: bases -> bf16 [S][M]; wbases -> bf16 transposed [M][S]
__global__ void prep_bases_kernel(const float* bases, const float* wbases,
                                  unsigned short* bases_bf, unsigned short* wbT) {
  int idx = blockIdx.x * blockDim.x + threadIdx.x;   // S*M threads
  int s = idx >> 6, k = idx & 63;
  bases_bf[idx] = f2bf(bases[idx]);
  wbT[(size_t)k * SDIM + s] = f2bf(wbases[idx]);
}

// fc0: h[b][c][s] = x[b][s][0]*w[0][c] + x[b][s][1]*w[1][c] + b[c]   (bf16 out)
__global__ void fc0_kernel(const float* x, const float* w, const float* bias, unsigned short* h) {
  int idx = blockIdx.x * blockDim.x + threadIdx.x;   // B*S threads
  int b = idx / SDIM, s = idx - b * SDIM;
  float x0 = x[(size_t)idx * 2 + 0];
  float x1 = x[(size_t)idx * 2 + 1];
  unsigned short* hb = h + (size_t)b * CDIM * SDIM + s;
#pragma unroll 8
  for (int c = 0; c < CDIM; ++c)
    hb[(size_t)c * SDIM] = f2bf(x0 * w[c] + x1 * w[CDIM + c] + bias[c]);
}

// Projection (split-K): part[b][chunk][i][k] = sum_{s in chunk} h[b][i][s] * wbT[k][s]
__global__ void proj_kernel(const unsigned short* __restrict__ h,
                            const unsigned short* __restrict__ wbT,
                            float* __restrict__ part) {
  int b  = blockIdx.x / NCHUNK;
  int ch = blockIdx.x - b * NCHUNK;
  int s_begin = ch * CHUNK;
  int wave = threadIdx.x >> 5;
  int t0 = wave * 2;
  int tr = t0 >> 2;           // tile row (channels)
  int tc0 = t0 & 3;           // tile cols (modes)
  int tc1 = tc0 + 1;
  const unsigned short* hb = h + (size_t)b * CDIM * SDIM;

  v8f acc0 = {}, acc1 = {};
  for (int s = s_begin; s < s_begin + CHUNK; s += 32) {
    __builtin_prefetch(hb + (size_t)(tr * 16) * SDIM + s + 256, 0, 1);
    v16bf a  = frag_a(hb,  tr * 16, s, SDIM);
    v16bf b0 = frag_b(wbT, tc0 * 16, s, SDIM);
    v16bf b1 = frag_b(wbT, tc1 * 16, s, SDIM);
    acc0 = wmma_bf16(a, b0, acc0);
    acc1 = wmma_bf16(a, b1, acc1);
  }
  float* pout = part + (size_t)(b * NCHUNK + ch) * CDIM * MDIM;
  store_c_f32(pout, tr * 16, tc0 * 16, MDIM, acc0);
  store_c_f32(pout, tr * 16, tc1 * 16, MDIM, acc1);
}

// Reduce split-K partials -> x_co[b][i][k] (f32)
__global__ void reduce_kernel(const float* __restrict__ part, float* __restrict__ xco) {
  int idx = blockIdx.x * blockDim.x + threadIdx.x;   // B*C*M threads
  int b = idx / (CDIM * MDIM);
  int r = idx - b * CDIM * MDIM;
  float s = 0.f;
  const float* p = part + (size_t)b * NCHUNK * CDIM * MDIM + r;
#pragma unroll 8
  for (int c = 0; c < NCHUNK; ++c) s += p[(size_t)c * CDIM * MDIM];
  xco[idx] = s;
}

// Banded spectral mixing: x_hat[b][o][k]  (bf16 out, ready as WMMA A operand)
__global__ void modemix_kernel(const float* __restrict__ xco,
                               const float* __restrict__ w0, const float* __restrict__ w1,
                               const float* __restrict__ w2, const float* __restrict__ w3,
                               const float* __restrict__ w4,
                               unsigned short* __restrict__ xhat) {
  int idx = blockIdx.x * blockDim.x + threadIdx.x;   // B*C*M threads
  int b = idx / (CDIM * MDIM);
  int r = idx - b * CDIM * MDIM;
  int o = r >> 6, k = r & 63;
  const float* xc = xco + (size_t)b * CDIM * MDIM;
  float acc = 0.f;
#pragma unroll 4
  for (int i = 0; i < CDIM; ++i)
    acc += xc[i * MDIM + k] * w0[(i * CDIM + o) * MDIM + k];
  if (k < 63) {
    float a = 0.f;
#pragma unroll 4
    for (int i = 0; i < CDIM; ++i) a += xc[i * MDIM + k + 1] * w1[(i * CDIM + o) * 63 + k];
    acc += a;
  }
  if (k >= 1) {
    float a = 0.f;
#pragma unroll 4
    for (int i = 0; i < CDIM; ++i) a += xc[i * MDIM + k - 1] * w2[(i * CDIM + o) * 63 + (k - 1)];
    acc += a;
  }
  if (k < 62) {
    float a = 0.f;
#pragma unroll 4
    for (int i = 0; i < CDIM; ++i) a += xc[i * MDIM + k + 2] * w3[(i * CDIM + o) * 62 + k];
    acc += a;
  }
  if (k >= 2) {
    float a = 0.f;
#pragma unroll 4
    for (int i = 0; i < CDIM; ++i) a += xc[i * MDIM + k - 2] * w4[(i * CDIM + o) * 62 + (k - 2)];
    acc += a;
  }
  xhat[idx] = f2bf(acc);
}

// Fused: res = x_hat·basesᵀ + Wᵀ·h + bias ; (gelu) ; h += res
// h tile and bases tile staged via TDM (tensor_load_to_lds, TENSORcnt).
__global__ void fused_layer_kernel(unsigned short* __restrict__ h,
                                   const unsigned short* __restrict__ xhat,
                                   const unsigned short* __restrict__ bases_bf,
                                   const unsigned short* __restrict__ wT,
                                   const float* __restrict__ wbias, int do_gelu) {
  __shared__ __align__(16) unsigned short hRow[64 * 64];   // [c][s] straight from TDM
  __shared__ __align__(16) unsigned short ldsB[64 * 72];   // [s][k] TDM-padded rows
  __shared__ __align__(16) unsigned short hT[64 * 72];     // [s][c] transposed
  int b  = blockIdx.x / STILES;
  int s0 = (blockIdx.x - b * STILES) * 64;
  unsigned short* hb = h + (size_t)b * CDIM * SDIM;

  if (threadIdx.x < 32) {
    // h tile: 64 rows (c) x 64 cols (s), row stride S elements, no LDS pad
    tdm_load_2d((unsigned)(uintptr_t)hRow, hb + s0, 64, 64, 64, 64,
                (unsigned long long)SDIM, 0, 0, 0);
    // bases tile: 64 rows (s) x 64 cols (k), contiguous; pad LDS rows 32dw+4dw = 72 shorts
    tdm_load_2d((unsigned)(uintptr_t)ldsB, bases_bf + (size_t)s0 * MDIM, 64, 64, 64, 64,
                (unsigned long long)MDIM, 1, 4, 3);
    __builtin_amdgcn_s_wait_tensorcnt(0);
  }
  __syncthreads();

  // transpose h tile in LDS for the x2 B-operand (and residual reads)
  for (int i = threadIdx.x; i < 64 * 64; i += 256) {
    int c = i >> 6, s = i & 63;
    hT[s * 72 + c] = hRow[c * 64 + s];
  }
  __syncthreads();

  int wave = threadIdx.x >> 5;
  int t0 = wave * 2;
  int tr = t0 >> 2;
  int tc0 = t0 & 3, tc1 = tc0 + 1;
  const unsigned short* xh = xhat + (size_t)b * CDIM * MDIM;

  v8f acc0 = {}, acc1 = {};
#pragma unroll
  for (int kk = 0; kk < MDIM; kk += 32) {   // x1: modes contraction (B from LDS bases)
    v16bf a  = frag_a(xh, tr * 16, kk, MDIM);
    v16bf b0 = frag_b(ldsB, tc0 * 16, kk, 72);
    v16bf b1 = frag_b(ldsB, tc1 * 16, kk, 72);
    acc0 = wmma_bf16(a, b0, acc0);
    acc1 = wmma_bf16(a, b1, acc1);
  }
#pragma unroll
  for (int kk = 0; kk < CDIM; kk += 32) {   // x2: channel contraction (B from LDS hT)
    v16bf a  = frag_a(wT, tr * 16, kk, CDIM);
    v16bf b0 = frag_b(hT, tc0 * 16, kk, 72);
    v16bf b1 = frag_b(hT, tc1 * 16, kk, 72);
    acc0 = wmma_bf16(a, b0, acc0);
    acc1 = wmma_bf16(a, b1, acc1);
  }

  int lane = threadIdx.x & 31;
  int radd = (lane >> 4) << 3;
#pragma unroll
  for (int t = 0; t < 2; ++t) {
    v8f acc = t ? acc1 : acc0;
    int tc = t ? tc1 : tc0;
    int n = tc * 16 + (lane & 15);
#pragma unroll
    for (int j = 0; j < 8; ++j) {
      int o = tr * 16 + radd + j;
      float res = acc[j] + wbias[o];
      if (do_gelu) res = gelu_exact(res);
      float hv = bf2f(hT[n * 72 + o]);
      hb[(size_t)o * SDIM + s0 + n] = f2bf(hv + res);
    }
  }
}

// Head: out[b][s] = fc2ᵀ · gelu(fc1ᵀ·h + b1) + b2   (h tile staged via TDM)
__global__ void head_kernel(const unsigned short* __restrict__ h,
                            const unsigned short* __restrict__ fc1T,
                            const float* __restrict__ fc1b,
                            const float* __restrict__ fc2w,
                            const float* __restrict__ fc2b,
                            float* __restrict__ out) {
  __shared__ __align__(16) unsigned short hRow[64 * 64];
  __shared__ __align__(16) unsigned short hT[64 * 72];
  __shared__ float oacc[64];
  int b  = blockIdx.x / STILES;
  int s0 = (blockIdx.x - b * STILES) * 64;
  const unsigned short* hb = h + (size_t)b * CDIM * SDIM;

  if (threadIdx.x < 32) {
    tdm_load_2d((unsigned)(uintptr_t)hRow, hb + s0, 64, 64, 64, 64,
                (unsigned long long)SDIM, 0, 0, 0);
    __builtin_amdgcn_s_wait_tensorcnt(0);
  }
  if (threadIdx.x >= 32 && threadIdx.x < 96) oacc[threadIdx.x - 32] = 0.f;
  __syncthreads();

  for (int i = threadIdx.x; i < 64 * 64; i += 256) {
    int c = i >> 6, s = i & 63;
    hT[s * 72 + c] = hRow[c * 64 + s];
  }
  __syncthreads();

  int wave = threadIdx.x >> 5;     // wave owns f-rows [wave*16, wave*16+16)
  int lane = threadIdx.x & 31;
  int radd = (lane >> 4) << 3;
#pragma unroll
  for (int tc = 0; tc < 4; ++tc) {
    v8f acc = {};
#pragma unroll
    for (int kk = 0; kk < CDIM; kk += 32) {
      v16bf a  = frag_a(fc1T, wave * 16, kk, CDIM);
      v16bf bf = frag_b(hT, tc * 16, kk, 72);
      acc = wmma_bf16(a, bf, acc);
    }
    int n = tc * 16 + (lane & 15);
    float p = 0.f;
#pragma unroll
    for (int j = 0; j < 8; ++j) {
      int f = wave * 16 + radd + j;
      float t = gelu_exact(acc[j] + fc1b[f]);
      p += t * fc2w[f];
    }
    atomicAdd(&oacc[n], p);
  }
  __syncthreads();
  if (threadIdx.x < 64)
    out[(size_t)b * SDIM + s0 + threadIdx.x] = oacc[threadIdx.x] + fc2b[0];
}

// ---------------------------------------------------------------------------
extern "C" void kernel_launch(void* const* d_in, const int* in_sizes, int n_in,
                              void* d_out, int out_size, void* d_ws, size_t ws_size,
                              hipStream_t stream) {
  // input index map (setup_inputs dict order, nested lists flattened)
  const float* x      = (const float*)d_in[0];
  const float* bases  = (const float*)d_in[1];
  const float* wbases = (const float*)d_in[2];
  const float* sp[NLAYERS][5];
  for (int l = 0; l < NLAYERS; ++l)
    for (int j = 0; j < 5; ++j) sp[l][j] = (const float*)d_in[3 + l * 5 + j];
  const float* w_w[NLAYERS], *w_b[NLAYERS];
  for (int l = 0; l < NLAYERS; ++l) {
    w_w[l] = (const float*)d_in[23 + l];
    w_b[l] = (const float*)d_in[27 + l];
  }
  const float* fc0w = (const float*)d_in[31];
  const float* fc0b = (const float*)d_in[32];
  const float* fc1w = (const float*)d_in[33];
  const float* fc1b = (const float*)d_in[34];
  const float* fc2w = (const float*)d_in[35];
  const float* fc2b = (const float*)d_in[36];
  float* out = (float*)d_out;

  // workspace carve-up
  char* ws = (char*)d_ws;
  size_t off = 0;
  auto carve = [&](size_t bytes) { char* p = ws + off; off += (bytes + 255) & ~(size_t)255; return p; };
  unsigned short* h        = (unsigned short*)carve((size_t)BDIM * CDIM * SDIM * 2);
  unsigned short* wbT      = (unsigned short*)carve((size_t)MDIM * SDIM * 2);
  unsigned short* bases_bf = (unsigned short*)carve((size_t)SDIM * MDIM * 2);
  float*          part     = (float*)carve((size_t)BDIM * NCHUNK * CDIM * MDIM * 4);
  float*          xco      = (float*)carve((size_t)BDIM * CDIM * MDIM * 4);
  unsigned short* xhat     = (unsigned short*)carve((size_t)BDIM * CDIM * MDIM * 2);
  unsigned short* wT       = (unsigned short*)carve((size_t)NLAYERS * CDIM * CDIM * 2);
  unsigned short* fc1T     = (unsigned short*)carve((size_t)FCDIM * CDIM * 2);
  (void)ws_size; (void)in_sizes; (void)n_in; (void)out_size;

  prep_weights_kernel<<<64, 256, 0, stream>>>(w_w[0], w_w[1], w_w[2], w_w[3], fc1w, wT, fc1T);
  prep_bases_kernel<<<(SDIM * MDIM) / 256, 256, 0, stream>>>(bases, wbases, bases_bf, wbT);
  fc0_kernel<<<(BDIM * SDIM) / 256, 256, 0, stream>>>(x, fc0w, fc0b, h);

  for (int l = 0; l < NLAYERS; ++l) {
    proj_kernel<<<BDIM * NCHUNK, 256, 0, stream>>>(h, wbT, part);
    reduce_kernel<<<(BDIM * CDIM * MDIM) / 256, 256, 0, stream>>>(part, xco);
    modemix_kernel<<<(BDIM * CDIM * MDIM) / 256, 256, 0, stream>>>(
        xco, sp[l][0], sp[l][1], sp[l][2], sp[l][3], sp[l][4], xhat);
    fused_layer_kernel<<<BDIM * STILES, 256, 0, stream>>>(
        h, xhat, bases_bf, wT + (size_t)l * CDIM * CDIM, w_b[l], (l != NLAYERS - 1) ? 1 : 0);
  }

  head_kernel<<<BDIM * STILES, 256, 0, stream>>>(h, fc1T, fc1b, fc2w, fc2b, out);
}
